// CoBiMambaLayer_77206332113397
// MI455X (gfx1250) — matrix operand order repaired and checked by reading, not compiled, not run on hardware
//
#include <hip/hip_runtime.h>

#define BB     8
#define LSEQ   4096
#define DMODEL 256
#define DINNER 512
#define DSTATE 16
#define DTRANK 16
#define DCONV  4
#define MROWS  (BB * LSEQ)   /* 32768 */
#define XDCOLS (DTRANK + 2 * DSTATE)  /* 48 */

typedef __attribute__((ext_vector_type(16))) __bf16         v16bf;
typedef __attribute__((ext_vector_type(16))) unsigned short v16us;
typedef __attribute__((ext_vector_type(8)))  unsigned short v8us;
typedef __attribute__((ext_vector_type(8)))  float          v8f;

static __device__ __forceinline__ unsigned short f32_to_bf16(float f) {
    unsigned int u = __float_as_uint(f);
    u += 0x7FFFu + ((u >> 16) & 1u);   // round-to-nearest-even
    return (unsigned short)(u >> 16);
}

static __device__ __forceinline__ v16bf cat_bf16(v8us lo, v8us hi) {
    v16us u = __builtin_shufflevector(lo, hi, 0, 1, 2, 3, 4, 5, 6, 7,
                                              8, 9, 10, 11, 12, 13, 14, 15);
    return __builtin_bit_cast(v16bf, u);
}

// ---------------------------------------------------------------- conversions
__global__ void cvt_bf16_kernel(const float* __restrict__ s,
                                unsigned short* __restrict__ d, long n) {
    long i = (long)blockIdx.x * blockDim.x + threadIdx.x;
    if (i < n) d[i] = f32_to_bf16(s[i]);
}

__global__ void neg_exp_kernel(const float* __restrict__ s,
                               float* __restrict__ d, int n) {
    int i = blockIdx.x * blockDim.x + threadIdx.x;
    if (i < n) d[i] = -__expf(s[i]);
}

// ---------------------------------------------------------------- WMMA GEMM
// C[M,N] (f32, row major, ldc) = A[M,K] (bf16 rm, lda) x Bw[N,K]^T (bf16 rm, ldb)
// One wave -> NT adjacent 16x16 output tiles along N (A loaded once per chunk).
// Main loop: K multiple-of-32 chunks, unconditional b128 vector loads.
// Epilogue: one half chunk when (K & 16) — used by the K=16 dt-projection.
template <int NT>
__global__ __launch_bounds__(256)
void wmma_gemm_bf16_kernel(const unsigned short* __restrict__ A, int lda,
                           const unsigned short* __restrict__ Bw, int ldb,
                           float* __restrict__ C, int ldc,
                           int M, int N, int K) {
    int wave  = (int)((blockIdx.x * blockDim.x + threadIdx.x) >> 5);
    int lane  = threadIdx.x & 31;
    int ng    = N / (16 * NT);          // tile groups along N
    int tg    = wave % ng;
    int tm    = wave / ng;
    if (tm * 16 >= M) return;           // wave-uniform -> EXEC stays all-ones

    int lmod  = lane & 15;
    int lhalf = lane >> 4;
    const unsigned short* Ap = A + (size_t)(tm * 16 + lmod) * lda;
    const unsigned short* Bp[NT];
#pragma unroll
    for (int t = 0; t < NT; ++t)
        Bp[t] = Bw + (size_t)(tg * NT * 16 + t * 16 + lmod) * ldb;

    v8f acc[NT];
#pragma unroll
    for (int t = 0; t < NT; ++t) acc[t] = (v8f){0.f,0.f,0.f,0.f,0.f,0.f,0.f,0.f};

    const int aoff = lhalf * 16;        // A: slots i    -> k = kb+aoff+i
    const int boff = lhalf * 16;        // B: slots i    -> k = kb+boff+i
    int kfull = K & ~31;

    for (int kb = 0; kb < kfull; kb += 32) {
        v8us a0 = *(const v8us*)(Ap + kb + aoff);
        v8us a1 = *(const v8us*)(Ap + kb + aoff + 8);
        v16bf av = cat_bf16(a0, a1);
#pragma unroll
        for (int t = 0; t < NT; ++t) {
            v8us b0 = *(const v8us*)(Bp[t] + kb + boff);
            v8us b1 = *(const v8us*)(Bp[t] + kb + boff + 8);
            acc[t] = __builtin_amdgcn_wmma_f32_16x16x32_bf16(
                         false, av, false, cat_bf16(b0, b1),
                         (short)0, acc[t], false, false);
        }
    }

    if (K & 16) {                       // trailing half chunk (dt-proj K=16)
        v8us zero = (v8us)0;
        v8us a0 = lhalf ? zero : *(const v8us*)(Ap + kfull);      // k = kfull..kfull+7
        v8us a1 = lhalf ? zero : *(const v8us*)(Ap + kfull + 8);  // k = kfull+8..kfull+15
        v16bf av = cat_bf16(a0, a1);                   // k >= kfull+16 -> 0
        v8us b0[NT], b1[NT];
        if (lhalf == 0) {               // only lanes with k in range load B
#pragma unroll
            for (int t = 0; t < NT; ++t) {
                b0[t] = *(const v8us*)(Bp[t] + kfull);
                b1[t] = *(const v8us*)(Bp[t] + kfull + 8);
            }
        } else {
#pragma unroll
            for (int t = 0; t < NT; ++t) { b0[t] = zero; b1[t] = zero; }
        }
#pragma unroll
        for (int t = 0; t < NT; ++t)
            acc[t] = __builtin_amdgcn_wmma_f32_16x16x32_bf16(
                         false, av, false, cat_bf16(b0[t], b1[t]),
                         (short)0, acc[t], false, false);
    }

#pragma unroll
    for (int t = 0; t < NT; ++t) {
        int ncol = tg * NT * 16 + t * 16 + lmod;
#pragma unroll
        for (int v = 0; v < 8; ++v) {   // C/D: VGPR v -> M = v + 8*lhalf
            int m = tm * 16 + lhalf * 8 + v;
            C[(size_t)m * ldc + ncol] = acc[t][v];
        }
    }
}

// ------------------------------------------------- depthwise causal conv+SiLU
__global__ void conv_silu_kernel(const float* __restrict__ xz,      // (M,1024) cols 0..511 = xi
                                 const float* __restrict__ w,       // (512,4)
                                 const float* __restrict__ bias,    // (512)
                                 float* __restrict__ xc,
                                 unsigned short* __restrict__ xc_bf) {
    long idx = (long)blockIdx.x * blockDim.x + threadIdx.x;  // m*512 + d
    if (idx >= (long)MROWS * DINNER) return;
    int d = (int)(idx & (DINNER - 1));
    long m = idx >> 9;
    int l = (int)(m & (LSEQ - 1));
    float acc = bias[d];
#pragma unroll
    for (int j = 0; j < DCONV; ++j) {
        int ls = l - (DCONV - 1) + j;
        if (ls >= 0)
            acc += xz[(size_t)(m - (DCONV - 1) + j) * (2 * DINNER) + d] * w[d * DCONV + j];
    }
    float sv = acc / (1.f + __expf(-acc));   // SiLU
    xc[idx]    = sv;
    xc_bf[idx] = f32_to_bf16(sv);
}

// -------------------------------------------------------- softplus(dt + bias)
__global__ void softplus_bias_kernel(float* __restrict__ dt,
                                     const float* __restrict__ dtb) {
    long idx = (long)blockIdx.x * blockDim.x + threadIdx.x;
    if (idx >= (long)MROWS * DINNER) return;
    int d = (int)(idx & (DINNER - 1));
    float x = dt[idx] + dtb[d];
    dt[idx] = (x > 20.f) ? x : log1pf(__expf(x));
}

// ---------------------------------------------------------- selective scan
// grid = BB * (DINNER/64), block = 64; thread owns channel d, h[16] in regs.
#define SCAN_T 64
__global__ __launch_bounds__(64)
void scan_kernel(const float* __restrict__ dt,      // (M,512)
                 const float* __restrict__ xc,      // (M,512)
                 const float* __restrict__ xz,      // (M,1024) cols 512.. = z
                 const float* __restrict__ xdbl,    // (M,48)  cols 16..31=B, 32..47=C
                 const float* __restrict__ Aneg,    // (512,16) = -exp(Alog)
                 const float* __restrict__ Dskip,   // (512)
                 float* __restrict__ y,
                 unsigned short* __restrict__ y_bf) {
    int b  = blockIdx.x >> 3;
    int dg = blockIdx.x & 7;
    int d  = dg * 64 + threadIdx.x;

    float Arow[DSTATE];
#pragma unroll
    for (int n = 0; n < DSTATE; ++n) Arow[n] = Aneg[d * DSTATE + n];
    float Dd = Dskip[d];
    float h[DSTATE];
#pragma unroll
    for (int n = 0; n < DSTATE; ++n) h[n] = 0.f;

    __shared__ float Bs[SCAN_T][DSTATE];
    __shared__ float Cs[SCAN_T][DSTATE];
    const size_t base = (size_t)b * LSEQ;

    for (int t0 = 0; t0 < LSEQ; t0 += SCAN_T) {
        __syncthreads();
        for (int i = threadIdx.x; i < SCAN_T * 32; i += blockDim.x) {
            int tt = i >> 5, c = i & 31;
            float v = xdbl[(base + t0 + tt) * XDCOLS + DTRANK + c];
            if (c < DSTATE) Bs[tt][c] = v; else Cs[tt][c - DSTATE] = v;
        }
        __syncthreads();
        for (int tt = 0; tt < SCAN_T; ++tt) {
            size_t m  = base + t0 + tt;
            float dtv = dt[m * DINNER + d];
            float u   = xc[m * DINNER + d];
            float dtu = dtv * u;
            float yv  = 0.f;
#pragma unroll
            for (int n = 0; n < DSTATE; ++n) {
                float dA = __expf(dtv * Arow[n]);
                h[n] = h[n] * dA + dtu * Bs[tt][n];
                yv  += h[n] * Cs[tt][n];
            }
            float zv  = xz[m * (2 * DINNER) + DINNER + d];
            float g   = yv + u * Dd;
            float out = g * (zv / (1.f + __expf(-zv)));   // * silu(z)
            y[m * DINNER + d]    = out;
            y_bf[m * DINNER + d] = f32_to_bf16(out);
        }
    }
}

// ----------------------------------------------------- LayerNorm + residual
__global__ __launch_bounds__(256)
void ln_residual_kernel(const float* __restrict__ xin,
                        const float* __restrict__ o1,
                        const float* __restrict__ w,
                        const float* __restrict__ bsf,
                        float* __restrict__ outp) {
    int m = blockIdx.x, t = threadIdx.x;
    float v = o1[(size_t)m * DMODEL + t];
    __shared__ float s1[DMODEL], s2[DMODEL];
    s1[t] = v; s2[t] = v * v;
    __syncthreads();
    for (int o = DMODEL / 2; o > 0; o >>= 1) {
        if (t < o) { s1[t] += s1[t + o]; s2[t] += s2[t + o]; }
        __syncthreads();
    }
    float mu  = s1[0] * (1.f / DMODEL);
    float var = s2[0] * (1.f / DMODEL) - mu * mu;
    float inv = rsqrtf(var + 1e-6f);
    outp[(size_t)m * DMODEL + t] =
        xin[(size_t)m * DMODEL + t] + (v - mu) * inv * w[t] + bsf[t];
}

// ---------------------------------------------------------------- launcher
extern "C" void kernel_launch(void* const* d_in, const int* in_sizes, int n_in,
                              void* d_out, int out_size, void* d_ws, size_t ws_size,
                              hipStream_t stream) {
    (void)in_sizes; (void)n_in; (void)out_size; (void)ws_size;
    char* ws = (char*)d_ws;
    size_t off = 0;
    auto alloc = [&](size_t bytes) -> size_t {
        size_t o = off; off += (bytes + 255) & ~(size_t)255; return o;
    };
    const size_t M = MROWS;
    size_t o_bfx   = alloc(M * DMODEL * 2);
    size_t o_bfinw = alloc((size_t)2 * DINNER * DMODEL * 2);
    size_t o_bfxp  = alloc((size_t)XDCOLS * DINNER * 2);
    size_t o_bfdtw = alloc((size_t)DINNER * DTRANK * 2);
    size_t o_bfow  = alloc((size_t)DMODEL * DINNER * 2);
    size_t o_An    = alloc((size_t)DINNER * DSTATE * 4);
    size_t o_xz    = alloc(M * 2 * DINNER * 4);
    size_t o_xc    = alloc(M * DINNER * 4);
    size_t o_bfxc  = alloc(M * DINNER * 2);
    size_t o_xdbl  = alloc(M * XDCOLS * 4);
    size_t o_bfxd  = alloc(M * XDCOLS * 2);
    size_t o_dt    = alloc(M * DINNER * 4);
    size_t o_y     = alloc(M * DINNER * 4);
    size_t o_bfy   = alloc(M * DINNER * 2);
    size_t o_o1    = alloc(M * DMODEL * 4);

    auto cvt = [&](const float* src, size_t dstoff, long n) {
        cvt_bf16_kernel<<<(unsigned)((n + 255) / 256), 256, 0, stream>>>(
            src, (unsigned short*)(ws + dstoff), n);
    };
    auto gemm = [&](size_t oa, int lda, size_t ob, int ldb, size_t oc, int ldc,
                    int Mm, int Nn, int Kk) {
        if (Nn % 64 == 0) {
            long waves = (long)(Mm / 16) * (Nn / 64);
            wmma_gemm_bf16_kernel<4><<<(unsigned)((waves + 7) / 8), 256, 0, stream>>>(
                (const unsigned short*)(ws + oa), lda,
                (const unsigned short*)(ws + ob), ldb,
                (float*)(ws + oc), ldc, Mm, Nn, Kk);
        } else if (Nn % 48 == 0) {
            long waves = (long)(Mm / 16) * (Nn / 48);
            wmma_gemm_bf16_kernel<3><<<(unsigned)((waves + 7) / 8), 256, 0, stream>>>(
                (const unsigned short*)(ws + oa), lda,
                (const unsigned short*)(ws + ob), ldb,
                (float*)(ws + oc), ldc, Mm, Nn, Kk);
        } else {
            long waves = (long)(Mm / 16) * (Nn / 16);
            wmma_gemm_bf16_kernel<1><<<(unsigned)((waves + 7) / 8), 256, 0, stream>>>(
                (const unsigned short*)(ws + oa), lda,
                (const unsigned short*)(ws + ob), ldb,
                (float*)(ws + oc), ldc, Mm, Nn, Kk);
        }
    };

    for (int s = 0; s < 2; ++s) {
        const float* xin   = (const float*)d_in[s];
        const float* in_w  = (const float*)d_in[2 + s * 9 + 0];
        const float* convw = (const float*)d_in[2 + s * 9 + 1];
        const float* convb = (const float*)d_in[2 + s * 9 + 2];
        const float* xpw   = (const float*)d_in[2 + s * 9 + 3];
        const float* dtw   = (const float*)d_in[2 + s * 9 + 4];
        const float* dtb   = (const float*)d_in[2 + s * 9 + 5];
        const float* Alog  = (const float*)d_in[2 + s * 9 + 6];
        const float* Dsk   = (const float*)d_in[2 + s * 9 + 7];
        const float* outw  = (const float*)d_in[2 + s * 9 + 8];
        const float* lnw   = (const float*)d_in[20 + s * 2];
        const float* lnb   = (const float*)d_in[21 + s * 2];
        float* outp = (float*)d_out + (size_t)s * M * DMODEL;

        // weight / activation conversions
        cvt(xin,  o_bfx,   (long)M * DMODEL);
        cvt(in_w, o_bfinw, 2L * DINNER * DMODEL);
        cvt(xpw,  o_bfxp,  (long)XDCOLS * DINNER);
        cvt(dtw,  o_bfdtw, (long)DINNER * DTRANK);
        cvt(outw, o_bfow,  (long)DMODEL * DINNER);
        neg_exp_kernel<<<(DINNER * DSTATE + 255) / 256, 256, 0, stream>>>(
            Alog, (float*)(ws + o_An), DINNER * DSTATE);

        // 1) in-projection: xz = x @ in_w.T   (cols 0..511 = xi, 512..1023 = z)
        gemm(o_bfx, DMODEL, o_bfinw, DMODEL, o_xz, 2 * DINNER,
             (int)M, 2 * DINNER, DMODEL);

        // 2) causal depthwise conv + SiLU
        long nconv = (long)M * DINNER;
        conv_silu_kernel<<<(unsigned)((nconv + 255) / 256), 256, 0, stream>>>(
            (const float*)(ws + o_xz), convw, convb,
            (float*)(ws + o_xc), (unsigned short*)(ws + o_bfxc));

        // 3) x-projection: x_dbl = xc @ xproj_w.T
        gemm(o_bfxc, DINNER, o_bfxp, DINNER, o_xdbl, XDCOLS,
             (int)M, XDCOLS, DINNER);
        cvt((const float*)(ws + o_xdbl), o_bfxd, (long)M * XDCOLS);

        // 4) dt-projection (K=16, zero padded) + softplus(dt + dt_b)
        gemm(o_bfxd, XDCOLS, o_bfdtw, DTRANK, o_dt, DINNER,
             (int)M, DINNER, DTRANK);
        softplus_bias_kernel<<<(unsigned)((nconv + 255) / 256), 256, 0, stream>>>(
            (float*)(ws + o_dt), dtb);

        // 5) selective scan (fused D-skip + SiLU(z) gating)
        scan_kernel<<<BB * (DINNER / 64), 64, 0, stream>>>(
            (const float*)(ws + o_dt), (const float*)(ws + o_xc),
            (const float*)(ws + o_xz), (const float*)(ws + o_xdbl),
            (const float*)(ws + o_An), Dsk,
            (float*)(ws + o_y), (unsigned short*)(ws + o_bfy));

        // 6) out-projection: o1 = y @ out_w.T
        gemm(o_bfy, DINNER, o_bfow, DINNER, o_o1, DMODEL,
             (int)M, DMODEL, DINNER);

        // 7) LayerNorm + residual
        ln_residual_kernel<<<(unsigned)M, 256, 0, stream>>>(
            xin, (const float*)(ws + o_o1), lnw, lnb, outp);
    }
}
